// CARM_1408749273689
// MI455X (gfx1250) — matrix operand order repaired and checked by reading, compiled, not run on hardware
//
#include <hip/hip_runtime.h>
#include <hip/hip_bf16.h>
#include <stdint.h>

typedef __attribute__((ext_vector_type(16))) __bf16       v16bf;
typedef __attribute__((ext_vector_type(8)))  float        v8f;
typedef __attribute__((ext_vector_type(8)))  unsigned int v8u;

#define NCH   10
#define FEAT  2048
#define OUTF  2048
#define BATCH 4096
#define MROWS (BATCH * NCH)      /* 40960 */
#define BN_EPS 1e-3f

#define BM     128               /* block tile M                 */
#define BN     256               /* block tile N                 */
#define BK     32                /* k-step                       */
#define NKSTEP (FEAT / BK)       /* 64                           */
#define LDSROW 80                /* bytes per LDS row (64 + 16 pad) */

/* LDS plane byte offsets within one buffer */
#define OFF_AH 0
#define OFF_AL (OFF_AH + BM * LDSROW)            /* 10240 */
#define OFF_BH (OFF_AL + BM * LDSROW)            /* 20480 */
#define OFF_BL (OFF_BH + BN * LDSROW)            /* 40960 */
#define BUFSZ  (OFF_BL + BN * LDSROW)            /* 61440 */

static __device__ __forceinline__ void split_bf16(float v, __bf16& h, __bf16& l) {
    h = (__bf16)v;
    l = (__bf16)(v - (float)h);
}

// ---------------------------------------------------------------------------
// Kernel 1: channel mixing with normalized adjacency + bf16 hi/lo split.
// ---------------------------------------------------------------------------
__global__ void __launch_bounds__(256) mix_split_kernel(
    const float* __restrict__ x, const float* __restrict__ adj,
    __bf16* __restrict__ xm_hi, __bf16* __restrict__ xm_lo)
{
    __shared__ float dinv[NCH];
    __shared__ float nadj[NCH][NCH];
    const int tid = threadIdx.x;

    if (tid < NCH) {
        float s = 0.f;
        #pragma unroll
        for (int k = 0; k < NCH; ++k) s += fabsf(adj[tid * NCH + k]);
        dinv[tid] = rsqrtf(fmaxf(s, 1e-8f));
    }
    __syncthreads();
    if (tid < NCH * NCH) {
        int i = tid / NCH, j = tid % NCH;
        nadj[i][j] = adj[i * NCH + j] * dinv[i] * dinv[j] + (i == j ? 1.0f : 0.0f);
    }
    __syncthreads();

    const int b  = blockIdx.x >> 2;
    const int f0 = ((blockIdx.x & 3) * 256 + tid) * 2;

    float2 v[NCH];
    #pragma unroll
    for (int j = 0; j < NCH; ++j)
        v[j] = *(const float2*)&x[((size_t)b * NCH + j) * FEAT + f0];

    unsigned int* ph = (unsigned int*)xm_hi;
    unsigned int* pl = (unsigned int*)xm_lo;
    #pragma unroll
    for (int i = 0; i < NCH; ++i) {
        float s0 = 0.f, s1 = 0.f;
        #pragma unroll
        for (int j = 0; j < NCH; ++j) {
            s0 = fmaf(nadj[i][j], v[j].x, s0);
            s1 = fmaf(nadj[i][j], v[j].y, s1);
        }
        __bf16 h0, l0, h1, l1;
        split_bf16(s0, h0, l0);
        split_bf16(s1, h1, l1);
        size_t o = (((size_t)b * NCH + i) * FEAT + f0) >> 1;
        ph[o] = (unsigned int)__builtin_bit_cast(unsigned short, h0) |
                ((unsigned int)__builtin_bit_cast(unsigned short, h1) << 16);
        pl[o] = (unsigned int)__builtin_bit_cast(unsigned short, l0) |
                ((unsigned int)__builtin_bit_cast(unsigned short, l1) << 16);
    }
}

// ---------------------------------------------------------------------------
// Kernel 2: weight transpose (O-major, F contiguous) + bf16 hi/lo split.
// ---------------------------------------------------------------------------
__global__ void __launch_bounds__(256) wsplit_kernel(
    const float* __restrict__ w,
    __bf16* __restrict__ wt_hi, __bf16* __restrict__ wt_lo)
{
    __shared__ float t[32][33];
    const int tx = threadIdx.x & 31, ty = threadIdx.x >> 5;
    const int o0 = blockIdx.x * 32, f0 = blockIdx.y * 32;
    #pragma unroll
    for (int i = 0; i < 4; ++i)
        t[ty + 8 * i][tx] = w[(size_t)(f0 + ty + 8 * i) * OUTF + (o0 + tx)];
    __syncthreads();
    #pragma unroll
    for (int i = 0; i < 4; ++i) {
        float v = t[tx][ty + 8 * i];
        __bf16 h, l;
        split_bf16(v, h, l);
        size_t o = (size_t)(o0 + ty + 8 * i) * FEAT + (f0 + tx);
        wt_hi[o] = h;
        wt_lo[o] = l;
    }
}

// ---------------------------------------------------------------------------
// Kernel 3: bf16x3 GEMM, async global->LDS double-buffered pipeline,
// fused bias+ELU+BN epilogue.
// Block 128x256, 8 waves (2 in M x 4 in N), wave tile 64x64 (4x4 WMMA tiles).
// ---------------------------------------------------------------------------
__global__ void __launch_bounds__(256) gemm_bf16x3_kernel(
    const __bf16* __restrict__ Ah, const __bf16* __restrict__ Al,
    const __bf16* __restrict__ Bh, const __bf16* __restrict__ Bl,
    const float* __restrict__ bias, const float* __restrict__ gamma,
    const float* __restrict__ beta, const float* __restrict__ mean,
    const float* __restrict__ var, float* __restrict__ out)
{
    __shared__ __align__(16) unsigned char lds[2 * BUFSZ];

    const int tid  = threadIdx.x;
    const int lane = tid & 31;
    const int wave = tid >> 5;
    const int wm   = wave & 1;            // 0..1 : wave row (64 rows)
    const int wn   = wave >> 1;           // 0..3 : wave col (64 cols)
    const int m0   = blockIdx.y * BM;
    const int n0   = blockIdx.x * BN;

    // ------------------ async-copy coordinates ------------------
    const int rr = tid >> 2;              // 0..63
    const int cc = tid & 3;               // 16B chunk in 64B row
    const unsigned lbase = (unsigned)(uintptr_t)lds;   // LDS byte address of buffer 0

    unsigned offA[2], dstA[2];            // shared by Ah/Al (same offsets, diff saddr)
    unsigned offB[4], dstB[4];
    #pragma unroll
    for (int p = 0; p < 2; ++p) {
        int row = rr + 64 * p;
        offA[p] = (unsigned)(m0 + row) * (FEAT * 2u) + (unsigned)cc * 16u;
        dstA[p] = lbase + (unsigned)(row * LDSROW + cc * 16);
    }
    #pragma unroll
    for (int q = 0; q < 4; ++q) {
        int row = rr + 64 * q;
        offB[q] = (unsigned)(n0 + row) * (FEAT * 2u) + (unsigned)cc * 16u;
        dstB[q] = lbase + (unsigned)(row * LDSROW + cc * 16);
    }

    auto issue_async = [&](int ks, int buf) {
        const unsigned bo   = buf ? (unsigned)BUFSZ : 0u;
        const unsigned gofs = (unsigned)ks * (BK * 2u);
        #pragma unroll
        for (int p = 0; p < 2; ++p) {
            unsigned ga = offA[p] + gofs;
            asm volatile("global_load_async_to_lds_b128 %0, %1, %2 offset:0"
                         :: "v"(dstA[p] + bo + OFF_AH), "v"(ga), "s"(Ah) : "memory");
            asm volatile("global_load_async_to_lds_b128 %0, %1, %2 offset:0"
                         :: "v"(dstA[p] + bo + OFF_AL), "v"(ga), "s"(Al) : "memory");
        }
        #pragma unroll
        for (int q = 0; q < 4; ++q) {
            unsigned gb = offB[q] + gofs;
            asm volatile("global_load_async_to_lds_b128 %0, %1, %2 offset:0"
                         :: "v"(dstB[q] + bo + OFF_BH), "v"(gb), "s"(Bh) : "memory");
            asm volatile("global_load_async_to_lds_b128 %0, %1, %2 offset:0"
                         :: "v"(dstB[q] + bo + OFF_BL), "v"(gb), "s"(Bl) : "memory");
        }
    };

    // ------------------ LDS fragment loads ------------------
    // A frag (16x32 bf16): lanes 0-15 K0..7|K16..23, lanes 16-31 K8..15|K24..31
    auto ldfragA = [&](const unsigned* s, int ms) -> v16bf {
        int row = wm * 64 + ms * 16 + (lane & 15);
        int off = row * LDSROW + ((lane >> 4) << 4);
        uint4 a = *(const uint4*)((const char*)s + off);
        uint4 b = *(const uint4*)((const char*)s + off + 32);
        v8u t;
        t[0] = a.x; t[1] = a.y; t[2] = a.z; t[3] = a.w;
        t[4] = b.x; t[5] = b.y; t[6] = b.z; t[7] = b.w;
        return __builtin_bit_cast(v16bf, t);
    };
    // B frag (32x16 bf16): lane = column; lanes 0-15 K0..7|K16..23, lanes 16-31 K8..15|K24..31
    auto ldfragB = [&](const unsigned* s, int ns) -> v16bf {
        int row = wn * 64 + ns * 16 + (lane & 15);
        int off = row * LDSROW + ((lane >> 4) << 4);
        uint4 a = *(const uint4*)((const char*)s + off);
        uint4 b = *(const uint4*)((const char*)s + off + 32);
        v8u t;
        t[0] = a.x; t[1] = a.y; t[2] = a.z; t[3] = a.w;
        t[4] = b.x; t[5] = b.y; t[6] = b.z; t[7] = b.w;
        return __builtin_bit_cast(v16bf, t);
    };

    v8f acc[4][4] = {};

    issue_async(0, 0);
    asm volatile("s_wait_asynccnt 0x0" ::: "memory");
    __syncthreads();

    #pragma unroll 1
    for (int ks = 0; ks < NKSTEP; ++ks) {
        const int  buf  = ks & 1;
        const bool more = (ks + 1) < NKSTEP;
        if (more) issue_async(ks + 1, buf ^ 1);

        const char* base = (const char*)lds + buf * BUFSZ;
        const unsigned* pAh = (const unsigned*)(base + OFF_AH);
        const unsigned* pAl = (const unsigned*)(base + OFF_AL);
        const unsigned* pBh = (const unsigned*)(base + OFF_BH);
        const unsigned* pBl = (const unsigned*)(base + OFF_BL);

        v16bf a_h[4], a_l[4];
        #pragma unroll
        for (int ms = 0; ms < 4; ++ms) {
            a_h[ms] = ldfragA(pAh, ms);
            a_l[ms] = ldfragA(pAl, ms);
        }
        #pragma unroll
        for (int ns = 0; ns < 4; ++ns) {
            v16bf b_h = ldfragB(pBh, ns);
            v16bf b_l = ldfragB(pBl, ns);
            #pragma unroll
            for (int ms = 0; ms < 4; ++ms) {
                acc[ms][ns] = __builtin_amdgcn_wmma_f32_16x16x32_bf16(
                    false, a_h[ms], false, b_h, (short)0, acc[ms][ns], false, false);
                acc[ms][ns] = __builtin_amdgcn_wmma_f32_16x16x32_bf16(
                    false, a_h[ms], false, b_l, (short)0, acc[ms][ns], false, false);
                acc[ms][ns] = __builtin_amdgcn_wmma_f32_16x16x32_bf16(
                    false, a_l[ms], false, b_h, (short)0, acc[ms][ns], false, false);
            }
        }
        asm volatile("s_wait_asynccnt 0x0" ::: "memory");
        __syncthreads();
    }

    // ------------------ epilogue: +bias -> ELU -> BN -> store ------------------
    float pb[4], ps[4], pf[4];
    int   pcol[4];
    #pragma unroll
    for (int ns = 0; ns < 4; ++ns) {
        int col  = n0 + wn * 64 + ns * 16 + (lane & 15);
        pcol[ns] = col;
        float sc = gamma[col] * rsqrtf(var[col] + BN_EPS);
        pb[ns]   = bias[col];
        ps[ns]   = sc;
        pf[ns]   = beta[col] - mean[col] * sc;
    }

    const int half = lane >> 4;
    #pragma unroll
    for (int ms = 0; ms < 4; ++ms) {
        #pragma unroll
        for (int ns = 0; ns < 4; ++ns) {
            #pragma unroll
            for (int r = 0; r < 8; ++r) {
                float vv = acc[ms][ns][r] + pb[ns];
                vv = vv > 0.f ? vv : (__expf(vv) - 1.0f);
                vv = fmaf(vv, ps[ns], pf[ns]);
                int rowg = m0 + wm * 64 + ms * 16 + half * 8 + r;
                out[(size_t)rowg * OUTF + pcol[ns]] = vv;
            }
        }
    }
}

// ---------------------------------------------------------------------------
extern "C" void kernel_launch(void* const* d_in, const int* in_sizes, int n_in,
                              void* d_out, int out_size, void* d_ws, size_t ws_size,
                              hipStream_t stream)
{
    const float* x     = (const float*)d_in[0];
    const float* adj   = (const float*)d_in[1];
    const float* w     = (const float*)d_in[2];
    const float* bias  = (const float*)d_in[3];
    const float* gamma = (const float*)d_in[4];
    const float* beta  = (const float*)d_in[5];
    const float* mean  = (const float*)d_in[6];
    const float* var   = (const float*)d_in[7];
    float* out = (float*)d_out;

    __bf16* xm_hi = (__bf16*)d_ws;
    __bf16* xm_lo = xm_hi + (size_t)MROWS * FEAT;
    __bf16* wt_hi = xm_lo + (size_t)MROWS * FEAT;
    __bf16* wt_lo = wt_hi + (size_t)FEAT * OUTF;

    mix_split_kernel<<<BATCH * (FEAT / 512), 256, 0, stream>>>(x, adj, xm_hi, xm_lo);
    wsplit_kernel<<<dim3(OUTF / 32, FEAT / 32), 256, 0, stream>>>(w, wt_hi, wt_lo);
    gemm_bf16x3_kernel<<<dim3(OUTF / BN, MROWS / BM), 256, 0, stream>>>(
        xm_hi, xm_lo, wt_hi, wt_lo, bias, gamma, beta, mean, var, out);
}
